// MultiHeadSelfAttention_58059367907442
// MI455X (gfx1250) — compile-verified, matrix operand order checked by reading, not run on hardware
//
#include <hip/hip_runtime.h>
#include <hip/hip_bf16.h>

// MHA forward for MI455X (gfx1250, wave32, WMMA f16 16x16x32, f32 accum).
// b=8, t=2048, d=512, h=8, dk=64.
// Data path: GLOBAL_LOAD_ASYNC_TO_LDS_B128 (ASYNCcnt) into double-buffered
// LDS tiles, software-pipelined against v_wmma_f32_16x16x32_f16.
// Workspace (~86 MB): xh, qh, kh, vT, ctxh (16 MiB each), WqkvT, WoutT.

#define B_   8
#define T_   2048
#define D_   512
#define H_   8
#define DK_  64
#define N3D_ 1536
#define MINVAL (-3.402823466e38f)

typedef __attribute__((ext_vector_type(16))) _Float16 v16h;
typedef __attribute__((ext_vector_type(8)))  _Float16 v8h;
typedef __attribute__((ext_vector_type(8)))  float    v8f;

// ---------------------------------------------------------------------------
// Async global -> LDS, 16 bytes per lane. GVS mode: addr = SGPR64 + VGPR32.
// vdst VGPR carries the per-lane LDS byte address (low 32 bits of a generic
// pointer into __shared__). Tracked by ASYNCcnt.
// ---------------------------------------------------------------------------
__device__ __forceinline__ void async_load_b128(void* lds_ptr, const void* gbase,
                                                unsigned byte_off) {
  unsigned lds_addr = (unsigned)(size_t)lds_ptr;
  asm volatile("global_load_async_to_lds_b128 %0, %1, %2"
               :
               : "v"(lds_addr), "v"(byte_off), "s"(gbase)
               : "memory");
}
__device__ __forceinline__ void wait_async_lds() {
  asm volatile("s_wait_asynccnt 0x0\n\ts_wait_dscnt 0x0" ::: "memory");
}

// ---------------------------------------------------------------------------
// Fragment loader for A (rows=M) and B (rows=N, LDS stored [n][k]) of
// v_wmma_f32_16x16x32_f16, per CDNA5 ISA 16-bit A layout:
//   lanes 0-15 : row = lane,    elems 0-7 = k 0..7,   elems 8-15 = k 16..23
//   lanes 16-31: row = lane-16, elems 0-7 = k 8..15,  elems 8-15 = k 24..31
// ---------------------------------------------------------------------------
__device__ __forceinline__ v16h load_frag(const _Float16* base, int strideH) {
  const int lane = threadIdx.x & 31;
  const int r    = lane & 15;
  const int hi   = lane >> 4;
  const _Float16* p = base + r * strideH + (hi << 3);
  v8h lo  = *(const v8h*)(p);
  v8h hi8 = *(const v8h*)(p + 16);
  v16h out;
#pragma unroll
  for (int i = 0; i < 8; ++i) { out[i] = lo[i]; out[i + 8] = hi8[i]; }
  return out;
}

__device__ __forceinline__ v8f wmma16(v16h a, v16h b, v8f c) {
  return __builtin_amdgcn_wmma_f32_16x16x32_f16(false, a, false, b,
                                                (short)0, c, false, false);
}

// ---------------------------------------------------------------------------
// Kernel 1: fp32 -> f16 conversions + weight transposes (to [N][K]).
// ---------------------------------------------------------------------------
__global__ __launch_bounds__(256) void cvt_kernel(
    const float* __restrict__ x, const float* __restrict__ Wqkv,
    const float* __restrict__ Wout, _Float16* __restrict__ xh,
    _Float16* __restrict__ WqkvT, _Float16* __restrict__ WoutT) {
  int idx = blockIdx.x * 256 + threadIdx.x;
  const int NX = B_ * T_ * D_;
  if (idx < NX) xh[idx] = (_Float16)x[idx];
  const int NW = N3D_ * D_;
  if (idx < NW) {
    int n = idx >> 9, k = idx & 511;
    WqkvT[idx] = (_Float16)Wqkv[k * N3D_ + n];
  }
  const int NO = D_ * D_;
  if (idx < NO) {
    int n = idx >> 9, k = idx & 511;
    WoutT[idx] = (_Float16)Wout[k * D_ + n];
  }
}

// ---------------------------------------------------------------------------
// Pipelined GEMM tile stage: async-load a 128x32-half A tile and B tile.
// ---------------------------------------------------------------------------
__device__ __forceinline__ void gemm_issue_tiles(
    _Float16* ldsA, _Float16* ldsB, const _Float16* gA, const _Float16* gB,
    int bm, int bn, int kk, int tid) {
#pragma unroll
  for (int i = 0; i < 2; ++i) {
    int c = tid + i * 256;                  // 512 chunks of 8 halves
    int row = c >> 2, col = (c & 3) * 8;
    async_load_b128(&ldsA[row * 40 + col], gA,
                    (unsigned)(((bm * 128 + row) * D_ + kk + col) * 2));
    async_load_b128(&ldsB[row * 40 + col], gB,
                    (unsigned)(((bn * 128 + row) * D_ + kk + col) * 2));
  }
}

// ---------------------------------------------------------------------------
// Kernel 2: QKV GEMM. C(16384x1536) = xh @ WqkvT^T + bqkv, scattered into
// head-major Q (scaled by dk^-0.5), K, and transposed V.
// 128x128x32 tiles, 8 waves (4Mx2N), 32x64 per wave; double-buffered LDS.
// ---------------------------------------------------------------------------
__global__ __launch_bounds__(256) void qkv_gemm_kernel(
    const _Float16* __restrict__ xh, const _Float16* __restrict__ WqkvT,
    const float* __restrict__ bqkv, _Float16* __restrict__ qh,
    _Float16* __restrict__ kh, _Float16* __restrict__ vT) {
  __shared__ __align__(16) _Float16 ldsA[2][128 * 40];
  __shared__ __align__(16) _Float16 ldsB[2][128 * 40];

  const int tid = threadIdx.x;
  const int wid = tid >> 5;
  const int wm  = wid & 3;
  const int wn  = wid >> 2;
  const int bm  = blockIdx.x / 12;
  const int bn  = blockIdx.x % 12;

  v8f acc[2][4];
#pragma unroll
  for (int i = 0; i < 2; ++i)
#pragma unroll
    for (int j = 0; j < 4; ++j) acc[i][j] = (v8f)(0.f);

  gemm_issue_tiles(ldsA[0], ldsB[0], xh, WqkvT, bm, bn, 0, tid);
  for (int step = 0; step < 16; ++step) {
    wait_async_lds();
    __syncthreads();
    if (step < 15)
      gemm_issue_tiles(ldsA[(step + 1) & 1], ldsB[(step + 1) & 1], xh, WqkvT,
                       bm, bn, (step + 1) * 32, tid);
    const _Float16* cA = ldsA[step & 1];
    const _Float16* cB = ldsB[step & 1];
    v16h a[2], b[4];
#pragma unroll
    for (int mi = 0; mi < 2; ++mi)
      a[mi] = load_frag(&cA[(wm * 32 + mi * 16) * 40], 40);
#pragma unroll
    for (int ni = 0; ni < 4; ++ni)
      b[ni] = load_frag(&cB[(wn * 64 + ni * 16) * 40], 40);
#pragma unroll
    for (int mi = 0; mi < 2; ++mi)
#pragma unroll
      for (int ni = 0; ni < 4; ++ni)
        acc[mi][ni] = wmma16(a[mi], b[ni], acc[mi][ni]);
  }

  const int lane = tid & 31, n_l = lane & 15, hi = lane >> 4;
#pragma unroll
  for (int mi = 0; mi < 2; ++mi) {
#pragma unroll
    for (int ni = 0; ni < 4; ++ni) {
#pragma unroll
      for (int i = 0; i < 8; ++i) {
        int grow = bm * 128 + wm * 32 + mi * 16 + i + 8 * hi;
        int gcol = bn * 128 + wn * 64 + ni * 16 + n_l;
        float val = acc[mi][ni][i] + bqkv[gcol];
        int sel = gcol >> 9, c0 = gcol & 511;
        int h = c0 >> 6, dkc = c0 & 63;
        int bb = grow >> 11, t = grow & 2047;
        if (sel == 0) {
          qh[((bb * H_ + h) * T_ + t) * DK_ + dkc] = (_Float16)(val * 0.125f);
        } else if (sel == 1) {
          kh[((bb * H_ + h) * T_ + t) * DK_ + dkc] = (_Float16)val;
        } else {
          vT[((bb * H_ + h) * DK_ + dkc) * T_ + t] = (_Float16)val;
        }
      }
    }
  }
}

// ---------------------------------------------------------------------------
// Kernel 3: flash attention. 1 workgroup = (b,h, 128 queries); 8 waves x 16
// query rows. Key blocks of 64, double-buffered async K/V, online softmax.
// ---------------------------------------------------------------------------
__device__ __forceinline__ void attn_issue_kv(
    _Float16* kdst, _Float16* vdst, const _Float16* kbase,
    const _Float16* vbase, int kstart, int tid) {
#pragma unroll
  for (int i = 0; i < 2; ++i) {
    int c = tid + i * 256;                  // 512 chunks of 8 halves
    int row = c >> 3, col = (c & 7) * 8;
    async_load_b128(&kdst[row * 72 + col], kbase,
                    (unsigned)(((kstart + row) * DK_ + col) * 2));
    async_load_b128(&vdst[row * 72 + col], vbase,
                    (unsigned)((row * T_ + kstart + col) * 2));
  }
}

__global__ __launch_bounds__(256) void attn_kernel(
    const _Float16* __restrict__ qh, const _Float16* __restrict__ kh,
    const _Float16* __restrict__ vT, const int* __restrict__ mask,
    _Float16* __restrict__ ctxh) {
  __shared__ __align__(16) _Float16 kbuf[2][64 * 72];   // [key][dk]
  __shared__ __align__(16) _Float16 vbuf[2][64 * 72];   // [dk][key]
  __shared__ __align__(16) _Float16 pbuf[128 * 72];     // [query][key]

  const int bh = blockIdx.x >> 4;
  const int qb = blockIdx.x & 15;
  const int b  = bh >> 3, h = bh & 7;
  const int tid = threadIdx.x, wid = tid >> 5;
  const int lane = tid & 31, n_l = lane & 15, hi = lane >> 4;
  const int qbase = qb * 128;

  const _Float16* qptr = qh + ((size_t)(b * H_ + h) * T_ + qbase + wid * 16) * DK_;
  v16h aq0 = load_frag(qptr, DK_);
  v16h aq1 = load_frag(qptr + 32, DK_);

  v8f o[4];
#pragma unroll
  for (int d = 0; d < 4; ++d) o[d] = (v8f)(0.f);
  float m_run[8], l_run[8];
#pragma unroll
  for (int r = 0; r < 8; ++r) { m_run[r] = MINVAL; l_run[r] = 0.f; }

  const _Float16* kbase = kh + (size_t)(b * H_ + h) * T_ * DK_;
  const _Float16* vbase = vT + (size_t)(b * H_ + h) * DK_ * T_;
  const int* mbase = mask + b * T_;

  attn_issue_kv(kbuf[0], vbuf[0], kbase, vbase, 0, tid);
  for (int kb = 0; kb < T_ / 64; ++kb) {
    const int kstart = kb * 64;
    wait_async_lds();
    __syncthreads();
    if (kb + 1 < T_ / 64)
      attn_issue_kv(kbuf[(kb + 1) & 1], vbuf[(kb + 1) & 1], kbase, vbase,
                    kstart + 64, tid);
    const _Float16* kcur = kbuf[kb & 1];
    const _Float16* vcur = vbuf[kb & 1];

    // S = Q * K^T : 4 column tiles x 2 k-steps.
    v8f s[4];
#pragma unroll
    for (int c = 0; c < 4; ++c) {
      v16h kf0 = load_frag(&kcur[(c * 16) * 72], 72);
      v16h kf1 = load_frag(&kcur[(c * 16) * 72 + 32], 72);
      s[c] = wmma16(aq0, kf0, (v8f)(0.f));
      s[c] = wmma16(aq1, kf1, s[c]);
    }

    int cm[4];
#pragma unroll
    for (int c = 0; c < 4; ++c) cm[c] = mbase[kstart + c * 16 + n_l];

    // Online softmax, row reductions via 16-lane butterflies.
#pragma unroll
    for (int r = 0; r < 8; ++r) {
      float smx[4];
#pragma unroll
      for (int c = 0; c < 4; ++c) smx[c] = cm[c] ? s[c][r] : MINVAL;
      float rmax = fmaxf(fmaxf(smx[0], smx[1]), fmaxf(smx[2], smx[3]));
#pragma unroll
      for (int off = 1; off < 16; off <<= 1)
        rmax = fmaxf(rmax, __shfl_xor(rmax, off, 32));
      float mn = fmaxf(m_run[r], rmax);
      float scale = __expf(m_run[r] - mn);
      float psum = 0.f;
#pragma unroll
      for (int c = 0; c < 4; ++c) {
        float p = cm[c] ? __expf(smx[c] - mn) : 0.f;  // post-softmax mask = 0
        pbuf[(wid * 16 + r + 8 * hi) * 72 + c * 16 + n_l] = (_Float16)p;
        psum += p;
      }
#pragma unroll
      for (int off = 1; off < 16; off <<= 1)
        psum += __shfl_xor(psum, off, 32);
      l_run[r] = l_run[r] * scale + psum;
      m_run[r] = mn;
#pragma unroll
      for (int d = 0; d < 4; ++d) o[d][r] *= scale;
    }
    asm volatile("s_wait_dscnt 0x0" ::: "memory");  // per-wave P store->load

    // O += P * V : A from pbuf (own wave's rows), B from vbuf [dk][key].
    v16h pa0 = load_frag(&pbuf[(wid * 16) * 72], 72);
    v16h pa1 = load_frag(&pbuf[(wid * 16) * 72 + 32], 72);
#pragma unroll
    for (int d = 0; d < 4; ++d) {
      v16h vb0 = load_frag(&vcur[(d * 16) * 72], 72);
      v16h vb1 = load_frag(&vcur[(d * 16) * 72 + 32], 72);
      o[d] = wmma16(pa0, vb0, o[d]);
      o[d] = wmma16(pa1, vb1, o[d]);
    }
  }

  // Normalize and store ctx as f16 [b][t][h*64+dk].
#pragma unroll
  for (int r = 0; r < 8; ++r) {
    float inv = (l_run[r] > 0.f) ? (1.f / l_run[r]) : 0.f;
    int t = qbase + wid * 16 + r + 8 * hi;
#pragma unroll
    for (int d = 0; d < 4; ++d) {
      ctxh[((size_t)b * T_ + t) * D_ + h * 64 + d * 16 + n_l] =
          (_Float16)(o[d][r] * inv);
    }
  }
}

// ---------------------------------------------------------------------------
// Kernel 4: output projection. out(16384x512) = ctxh @ WoutT^T + bout, fp32.
// ---------------------------------------------------------------------------
__global__ __launch_bounds__(256) void out_gemm_kernel(
    const _Float16* __restrict__ ctxh, const _Float16* __restrict__ WoutT,
    const float* __restrict__ bout, float* __restrict__ out) {
  __shared__ __align__(16) _Float16 ldsA[2][128 * 40];
  __shared__ __align__(16) _Float16 ldsB[2][128 * 40];

  const int tid = threadIdx.x;
  const int wid = tid >> 5;
  const int wm  = wid & 3;
  const int wn  = wid >> 2;
  const int bm  = blockIdx.x >> 2;
  const int bn  = blockIdx.x & 3;

  v8f acc[2][4];
#pragma unroll
  for (int i = 0; i < 2; ++i)
#pragma unroll
    for (int j = 0; j < 4; ++j) acc[i][j] = (v8f)(0.f);

  gemm_issue_tiles(ldsA[0], ldsB[0], ctxh, WoutT, bm, bn, 0, tid);
  for (int step = 0; step < 16; ++step) {
    wait_async_lds();
    __syncthreads();
    if (step < 15)
      gemm_issue_tiles(ldsA[(step + 1) & 1], ldsB[(step + 1) & 1], ctxh, WoutT,
                       bm, bn, (step + 1) * 32, tid);
    const _Float16* cA = ldsA[step & 1];
    const _Float16* cB = ldsB[step & 1];
    v16h a[2], b[4];
#pragma unroll
    for (int mi = 0; mi < 2; ++mi)
      a[mi] = load_frag(&cA[(wm * 32 + mi * 16) * 40], 40);
#pragma unroll
    for (int ni = 0; ni < 4; ++ni)
      b[ni] = load_frag(&cB[(wn * 64 + ni * 16) * 40], 40);
#pragma unroll
    for (int mi = 0; mi < 2; ++mi)
#pragma unroll
      for (int ni = 0; ni < 4; ++ni)
        acc[mi][ni] = wmma16(a[mi], b[ni], acc[mi][ni]);
  }

  const int lane = tid & 31, n_l = lane & 15, hi = lane >> 4;
#pragma unroll
  for (int mi = 0; mi < 2; ++mi)
#pragma unroll
    for (int ni = 0; ni < 4; ++ni)
#pragma unroll
      for (int i = 0; i < 8; ++i) {
        int grow = bm * 128 + wm * 32 + mi * 16 + i + 8 * hi;
        int gcol = bn * 128 + wn * 64 + ni * 16 + n_l;
        out[(size_t)grow * D_ + gcol] = acc[mi][ni][i] + bout[gcol];
      }
}

// ---------------------------------------------------------------------------
extern "C" void kernel_launch(void* const* d_in, const int* in_sizes, int n_in,
                              void* d_out, int out_size, void* d_ws,
                              size_t ws_size, hipStream_t stream) {
  const float* x    = (const float*)d_in[0];
  const int*   mask = (const int*)d_in[1];
  const float* Wqkv = (const float*)d_in[2];
  const float* bqkv = (const float*)d_in[3];
  const float* Wout = (const float*)d_in[4];
  const float* bout = (const float*)d_in[5];
  float* out = (float*)d_out;

  char* ws = (char*)d_ws;
  const size_t SZ_XH = (size_t)B_ * T_ * D_ * 2;       // 16 MiB
  _Float16* xh    = (_Float16*)(ws);
  _Float16* qh    = (_Float16*)(ws + SZ_XH);
  _Float16* kh    = (_Float16*)(ws + 2 * SZ_XH);
  _Float16* vT    = (_Float16*)(ws + 3 * SZ_XH);
  _Float16* ctxh  = (_Float16*)(ws + 4 * SZ_XH);
  _Float16* WqkvT = (_Float16*)(ws + 5 * SZ_XH);
  _Float16* WoutT = (_Float16*)(ws + 5 * SZ_XH + (size_t)N3D_ * D_ * 2);

  cvt_kernel<<<(B_ * T_ * D_) / 256, 256, 0, stream>>>(x, Wqkv, Wout, xh,
                                                       WqkvT, WoutT);
  qkv_gemm_kernel<<<(B_ * T_ / 128) * (N3D_ / 128), 256, 0, stream>>>(
      xh, WqkvT, bqkv, qh, kh, vT);
  attn_kernel<<<B_ * H_ * (T_ / 128), 256, 0, stream>>>(qh, kh, vT, mask, ctxh);
  out_gemm_kernel<<<(B_ * T_ / 128) * (D_ / 128), 256, 0, stream>>>(
      ctxh, WoutT, bout, out);
}
